// CenterHead_50869592654994
// MI455X (gfx1250) — compile-verified, matrix operand order checked by reading, not run on hardware
//
#include <hip/hip_runtime.h>
#include <math.h>

#define HH_ 180
#define WW_ 180
#define HWP_ (HH_*WW_)
#define BB_ 4
#define MTOT_ (BB_*HWP_)   // 129600

typedef _Float16 half_t;
typedef __attribute__((ext_vector_type(16))) _Float16 v16h;
typedef __attribute__((ext_vector_type(8)))  float    v8f;

// ---------------------------------------------------------------------------
// NCHW f32 -> NHWC f16 tiled transpose (coalesced reads along W, writes along C)
// ---------------------------------------------------------------------------
__global__ __launch_bounds__(256)
void nchw_to_nhwc_f2h(const float* __restrict__ x, half_t* __restrict__ xh, int C) {
  __shared__ float tile[32][33];
  int bh = blockIdx.z;                 // b*H + h
  int c0 = blockIdx.y * 32;
  int w0 = blockIdx.x * 32;
  int b = bh / HH_, h = bh % HH_;
  int tj = threadIdx.x & 31;
  int ti = threadIdx.x >> 5;           // 0..7
  for (int r = 0; r < 32; r += 8) {
    int c = c0 + ti + r;
    int w = w0 + tj;
    float v = 0.f;
    if (w < WW_) v = x[(((size_t)b * C + c) * HH_ + h) * WW_ + w];
    tile[ti + r][tj] = v;
  }
  __syncthreads();
  for (int r = 0; r < 32; r += 8) {
    int w = w0 + ti + r;
    int c = c0 + tj;
    if (w < WW_)
      xh[((size_t)(b * HH_ + h) * WW_ + w) * C + c] = (half_t)tile[tj][ti + r];
  }
}

// ---------------------------------------------------------------------------
// Weight repack: [N][Cin][3][3] f32 -> [9][N][Cin] f16 (shared conv)
// ---------------------------------------------------------------------------
__global__ __launch_bounds__(256)
void prep_shared_w(const float* __restrict__ w, half_t* __restrict__ out) {
  int i = blockIdx.x * 256 + threadIdx.x;         // 9*64*512
  if (i >= 9 * 64 * 512) return;
  int ci = i % 512; int t = i / 512;
  int n = t % 64;   int khw = t / 64;
  int kh = khw / 3, kw = khw % 3;
  out[i] = (half_t)w[((n * 512 + ci) * 3 + kh) * 3 + kw];
}

struct TaskW  { const float* w1[5]; };
struct TaskBN { const float* b1[5]; const float* s1[5]; const float* be1[5]; };

// fused 5-head conv1 weights: [9][320][64] f16, n = head*64 + och
__global__ __launch_bounds__(256)
void prep_task_w(TaskW tw, half_t* __restrict__ out) {
  int i = blockIdx.x * 256 + threadIdx.x;         // 9*320*64
  if (i >= 9 * 320 * 64) return;
  int ci = i % 64; int t = i / 64;
  int n = t % 320; int khw = t / 320;
  int head = n / 64, och = n % 64;
  int kh = khw / 3, kw = khw % 3;
  out[i] = (half_t)tw.w1[head][((och * 64 + ci) * 3 + kh) * 3 + kw];
}

__global__ void prep_task_bn(TaskBN tb, float* __restrict__ scale, float* __restrict__ bias) {
  int n = blockIdx.x * 64 + threadIdx.x;
  if (n >= 320) return;
  int head = n / 64, och = n % 64;
  float s = tb.s1[head][och];
  scale[n] = s;
  bias[n]  = tb.b1[head][och] * s + tb.be1[head][och];
}

// ---------------------------------------------------------------------------
// async global -> LDS staging primitive (CDNA5, ASYNCcnt-tracked)
// ---------------------------------------------------------------------------
__device__ __forceinline__ void async_ld_b128(unsigned lds_off, const half_t* g) {
  asm volatile("global_load_async_to_lds_b128 %0, %1, off"
               :: "v"(lds_off), "v"((unsigned long long)(uintptr_t)g)
               : "memory");
}
__device__ __forceinline__ void wait_async0() {
  asm volatile("s_wait_asynccnt 0x0" ::: "memory");
}

// ---------------------------------------------------------------------------
// Implicit-GEMM 3x3 conv, WMMA f16->f32, fused BN+ReLU, NHWC f16 in/out.
// Block: 256 threads (8 waves). Tile: M=256 (spatial), N=64 (blockIdx.y*64).
// K streamed as 9 taps x (Cin/32) chunks; double-buffered LDS filled by
// global_load_async_to_lds_b128, overlapped with 8 WMMAs/wave per chunk.
// ---------------------------------------------------------------------------
__global__ __launch_bounds__(256)
void conv3x3_wmma(const half_t* __restrict__ src, int Cin,
                  const half_t* __restrict__ wgt, int Ntot,
                  const float* __restrict__ scale, const float* __restrict__ bias,
                  half_t* __restrict__ dst) {
  __shared__ __align__(16) half_t ldsA[2][256 * 40];   // rows padded to 80B
  __shared__ __align__(16) half_t ldsB[2][64 * 40];

  const int tid  = threadIdx.x;
  const int wave = tid >> 5;
  const int lane = tid & 31;
  const int m0 = blockIdx.x * 256;
  const int n0 = blockIdx.y * 64;

  // chunk-invariant staging coordinates (4 A-rows + 1 B-row per thread)
  const int segA = tid & 3;
  int ohs[4], ows[4], bases[4];
  bool vmv[4];
#pragma unroll
  for (int it = 0; it < 4; ++it) {
    int row = (tid >> 2) + it * 64;
    int m = m0 + row;
    vmv[it] = (m < MTOT_);
    int mm = vmv[it] ? m : 0;
    int b = mm / HWP_; int rem = mm - b * HWP_;
    ohs[it] = rem / WW_; ows[it] = rem - ohs[it] * WW_;
    bases[it] = ((b * HH_ + ohs[it]) * WW_ + ows[it]) * Cin + segA * 8;
  }
  const int nB = tid >> 2;
  const unsigned aOff[2] = { (unsigned)(uintptr_t)&ldsA[0][0],
                             (unsigned)(uintptr_t)&ldsA[1][0] };
  const unsigned bOff[2] = { (unsigned)(uintptr_t)&ldsB[0][0],
                             (unsigned)(uintptr_t)&ldsB[1][0] };

  const int chunksC = Cin >> 5;
  const int total = 9 * chunksC;

  auto stage = [&](int idx, int p) {
    int khw = idx / chunksC, cc = idx - khw * chunksC;
    int kh = khw / 3, kw = khw % 3;
    int tapOff = ((kh - 1) * WW_ + (kw - 1)) * Cin + cc * 32;
    half_t* aPtr = &ldsA[p][0];
#pragma unroll
    for (int it = 0; it < 4; ++it) {
      int row = (tid >> 2) + it * 64;
      int ih = ohs[it] + kh - 1, iw = ows[it] + kw - 1;
      unsigned lo = aOff[p] + (unsigned)(row * 40 + segA * 8) * 2u;
      if (vmv[it] && ih >= 0 && ih < HH_ && iw >= 0 && iw < WW_) {
        async_ld_b128(lo, src + bases[it] + tapOff);
      } else {
        *(float4*)(&aPtr[row * 40 + segA * 8]) = make_float4(0.f, 0.f, 0.f, 0.f);
      }
    }
    async_ld_b128(bOff[p] + (unsigned)(nB * 40 + segA * 8) * 2u,
                  wgt + ((size_t)(khw * Ntot + n0 + nB)) * Cin + cc * 32 + segA * 8);
  };

  v8f zero = {0.f,0.f,0.f,0.f,0.f,0.f,0.f,0.f};
  v8f acc[2][4];
#pragma unroll
  for (int r = 0; r < 2; ++r)
#pragma unroll
    for (int j = 0; j < 4; ++j) acc[r][j] = zero;

  stage(0, 0);
  wait_async0();
  __syncthreads();

  const int koff = (lane < 16) ? 0 : 8;
  const int lrow = lane & 15;

  for (int i = 0; i < total; ++i) {
    int p = i & 1;
    if (i + 1 < total) stage(i + 1, p ^ 1);

    const half_t* A  = &ldsA[p][0];
    const half_t* Bt = &ldsB[p][0];
    v16h af[2];
#pragma unroll
    for (int r = 0; r < 2; ++r) {
      int rowA = wave * 32 + r * 16 + lrow;
      union { float4 f[2]; v16h v; } u;
      u.f[0] = *(const float4*)(&A[rowA * 40 + koff]);
      u.f[1] = *(const float4*)(&A[rowA * 40 + koff + 16]);
      af[r] = u.v;
    }
#pragma unroll
    for (int j = 0; j < 4; ++j) {
      int rowB = j * 16 + lrow;
      union { float4 f[2]; v16h v; } u;
      u.f[0] = *(const float4*)(&Bt[rowB * 40 + koff]);
      u.f[1] = *(const float4*)(&Bt[rowB * 40 + koff + 16]);
      acc[0][j] = __builtin_amdgcn_wmma_f32_16x16x32_f16(
          false, af[0], false, u.v, (short)0, acc[0][j], false, false);
      acc[1][j] = __builtin_amdgcn_wmma_f32_16x16x32_f16(
          false, af[1], false, u.v, (short)0, acc[1][j], false, false);
    }
    wait_async0();
    __syncthreads();
  }

  // epilogue: C/D layout lane<16 -> M=q, lane>=16 -> M=q+8; N = 16j + lane%16
  const int ncol = lane & 15;
#pragma unroll
  for (int r = 0; r < 2; ++r) {
    int mbase = m0 + wave * 32 + r * 16 + ((lane < 16) ? 0 : 8);
#pragma unroll
    for (int j = 0; j < 4; ++j) {
      int n = n0 + j * 16 + ncol;
      float sc = scale[n], bi = bias[n];
#pragma unroll
      for (int q = 0; q < 8; ++q) {
        int m = mbase + q;
        if (m < MTOT_) {
          float v = acc[r][j][q] * sc + bi;
          v = v > 0.f ? v : 0.f;
          dst[(size_t)m * Ntot + n] = (half_t)v;
        }
      }
    }
  }
}

// ---------------------------------------------------------------------------
// Tiny second conv (N<=3): direct, f16 in (NHWC, channel offset), planar f32 out
// ---------------------------------------------------------------------------
__global__ __launch_bounds__(256)
void conv2_direct(const half_t* __restrict__ fmid, int Ctot, int coff,
                  const float* __restrict__ w2, const float* __restrict__ b2,
                  float* __restrict__ outp, int cout) {
  int m = blockIdx.x * 256 + threadIdx.x;
  if (m >= MTOT_) return;
  int b = m / HWP_; int rem = m - b * HWP_;
  int oh = rem / WW_; int ow = rem - oh * WW_;
  float acc[3] = {0.f, 0.f, 0.f};
  for (int kh = 0; kh < 3; ++kh) {
    int ih = oh + kh - 1; if (ih < 0 || ih >= HH_) continue;
    for (int kw = 0; kw < 3; ++kw) {
      int iw = ow + kw - 1; if (iw < 0 || iw >= WW_) continue;
      const half_t* f = fmid + (size_t)((b * HH_ + ih) * WW_ + iw) * Ctot + coff;
      for (int ci = 0; ci < 64; ++ci) {
        float fv = (float)f[ci];
        for (int c = 0; c < 3; ++c)
          if (c < cout) acc[c] += fv * w2[((c * 64 + ci) * 3 + kh) * 3 + kw];
      }
    }
  }
  for (int c = 0; c < 3; ++c)
    if (c < cout) outp[(size_t)c * MTOT_ + m] = acc[c] + b2[c];
}

// ---------------------------------------------------------------------------
// sigmoid + 3x3 max NMS (peak test on logits: sigmoid is monotonic)
// hm planes: [c][B*HW]; out: [(b*C+c)][HW]
// ---------------------------------------------------------------------------
__global__ __launch_bounds__(256)
void nms_sigmoid(const float* __restrict__ hm, int C, float* __restrict__ out) {
  int i = blockIdx.x * 256 + threadIdx.x;
  int tot = BB_ * C * HWP_;
  if (i >= tot) return;
  int p = i % HWP_; int t = i / HWP_;
  int b = t / C, c = t % C;
  const float* plane = hm + (size_t)c * MTOT_ + (size_t)b * HWP_;
  int oh = p / WW_, ow = p % WW_;
  float v = plane[p];
  bool peak = true;
  for (int dh = -1; dh <= 1; ++dh)
    for (int dw = -1; dw <= 1; ++dw) {
      int ih = oh + dh, iw = ow + dw;
      if (ih < 0 || ih >= HH_ || iw < 0 || iw >= WW_) continue;
      if (plane[ih * WW_ + iw] > v) peak = false;
    }
  out[(size_t)t * HWP_ + p] = peak ? 1.f / (1.f + expf(-v)) : 0.f;
}

__global__ void init_cand(float* __restrict__ cs, int* __restrict__ ci) {
  int i = blockIdx.x * 256 + threadIdx.x;
  if (i < BB_ * 2 * 1024) { cs[i] = -1.0f; ci[i] = 0; }
}

// ---------------------------------------------------------------------------
// per-(b,c) top-500 superset via 11-bit radix histogram threshold
// ---------------------------------------------------------------------------
__global__ __launch_bounds__(256)
void topk_select(const float* __restrict__ nms, int C,
                 float* __restrict__ cs, int* __restrict__ ci) {
  __shared__ unsigned hist[2048];
  __shared__ int sT, sHi, sEq;
  int blk = blockIdx.x;                  // b*C + c
  int b = blk / C, c = blk % C;
  const float* plane = nms + (size_t)blk * HWP_;
  int tid = threadIdx.x;
  for (int i = tid; i < 2048; i += 256) hist[i] = 0u;
  __syncthreads();
  for (int p = tid; p < HWP_; p += 256) {
    unsigned bin = __float_as_uint(plane[p]) >> 21;   // scores >= 0
    atomicAdd(&hist[bin], 1u);
  }
  __syncthreads();
  if (tid == 0) {
    unsigned cum = 0; int T = 0; int base = 0;
    for (int i = 2047; i >= 0; --i) {
      if (cum + hist[i] >= 500u) { T = i; base = (int)cum; break; }
      cum += hist[i];
    }
    sT = T; sEq = base; sHi = 0;
  }
  __syncthreads();
  int T = sT;
  float* mycs = cs + (size_t)(b * 2 + c) * 1024;
  int*   myci = ci + (size_t)(b * 2 + c) * 1024;
  for (int p = tid; p < HWP_; p += 256) {
    float v = plane[p];
    int bin = (int)(__float_as_uint(v) >> 21);
    if (bin > T) {
      int pos = atomicAdd(&sHi, 1);
      if (pos < 1024) { mycs[pos] = v; myci[pos] = p; }
    } else if (bin == T) {
      int pos = atomicAdd(&sEq, 1);
      if (pos < 1024) { mycs[pos] = v; myci[pos] = p; }
    }
  }
}

// ---------------------------------------------------------------------------
// per-batch bitonic sort (2048) of candidates, gather features, emit outputs
// out layout: scores[B*3000] | labels[B*3000] | boxes[B*3000*7]
// ---------------------------------------------------------------------------
__global__ __launch_bounds__(512)
void sort_and_emit(const float* __restrict__ cs, const int* __restrict__ ci,
                   const float* __restrict__ planes, int C, int task, int labelBase,
                   float* __restrict__ out) {
  __shared__ float    ss[2048];
  __shared__ unsigned sp[2048];
  int b = blockIdx.x;
  int tid = threadIdx.x;
  for (int i = tid; i < 2048; i += 512) {
    float v = cs[(size_t)b * 2048 + i];
    int idx = ci[(size_t)b * 2048 + i];
    int c = i >> 10;
    if (c >= C) v = -1.0f;
    ss[i] = v;
    sp[i] = ((unsigned)c << 16) | ((unsigned)idx & 0xFFFFu);
  }
  __syncthreads();
  for (int k = 2; k <= 2048; k <<= 1) {
    for (int j = k >> 1; j > 0; j >>= 1) {
      for (int i = tid; i < 2048; i += 512) {
        int ixj = i ^ j;
        if (ixj > i) {
          bool desc = ((i & k) == 0);
          float a = ss[i], bb = ss[ixj];
          if (desc ? (a < bb) : (a > bb)) {
            ss[i] = bb; ss[ixj] = a;
            unsigned tp = sp[i]; sp[i] = sp[ixj]; sp[ixj] = tp;
          }
        }
      }
      __syncthreads();
    }
  }
  if (tid < 500) {
    float score = ss[tid];
    unsigned pl = sp[tid];
    int c = (int)(pl >> 16);
    int p = (int)(pl & 0xFFFFu);
    if (score < 0.f) { score = 0.f; c = 0; p = 0; }
    int oh = p / WW_, ow = p % WW_;
    size_t bp = (size_t)b * HWP_ + p;
    float regx = planes[(size_t)2 * MTOT_ + bp];
    float regy = planes[(size_t)3 * MTOT_ + bp];
    float hei  = planes[(size_t)4 * MTOT_ + bp];
    float d0 = expf(planes[(size_t)5 * MTOT_ + bp]);
    float d1 = expf(planes[(size_t)6 * MTOT_ + bp]);
    float d2 = expf(planes[(size_t)7 * MTOT_ + bp]);
    float rs = planes[(size_t)8 * MTOT_ + bp];
    float rc = planes[(size_t)9 * MTOT_ + bp];
    float rot = atan2f(rs, rc);
    float xs = (float)ow + regx;
    float ys = (float)oh + regy;
    float X = xs * 0.6f - 54.0f;   // stride*voxel = 8*0.075
    float Y = ys * 0.6f - 54.0f;
    int kout = b * 3000 + task * 500 + tid;
    out[kout] = score;
    out[12000 + kout] = (float)(labelBase + c);
    float* bx = out + 24000 + (size_t)kout * 7;
    bx[0] = X; bx[1] = Y; bx[2] = hei; bx[3] = d0; bx[4] = d1; bx[5] = d2; bx[6] = rot;
  }
}

// ---------------------------------------------------------------------------
extern "C" void kernel_launch(void* const* d_in, const int* in_sizes, int n_in,
                              void* d_out, int out_size, void* d_ws, size_t ws_size,
                              hipStream_t stream) {
  const long XN = (long)BB_ * 512 * HH_ * WW_;
  const bool insertion = (n_in > 0 && in_sizes[0] == (int)XN);

  const float *x, *sw, *ssc, *sbi;
  const float *w1p[6][5], *b1p[6][5], *s1p[6][5], *be1p[6][5], *w2p[6][5], *b2p[6][5];
  // canonical head order: 0=reg,1=height,2=dim,3=rot,4=hm
  if (insertion) {
    x   = (const float*)d_in[0];
    sw  = (const float*)d_in[1];
    ssc = (const float*)d_in[2];
    sbi = (const float*)d_in[3];
    for (int t = 0; t < 6; ++t)
      for (int h = 0; h < 5; ++h) {
        void* const* a = d_in + 4 + (t * 5 + h) * 6;  // w1,b1,s1,be1,w2,b2
        w1p[t][h] = (const float*)a[0]; b1p[t][h]  = (const float*)a[1];
        s1p[t][h] = (const float*)a[2]; be1p[t][h] = (const float*)a[3];
        w2p[t][h] = (const float*)a[4]; b2p[t][h]  = (const float*)a[5];
      }
  } else {
    // jax pytree sorted keys: shared_bias, shared_scale, shared_w, tasks..., x
    sbi = (const float*)d_in[0];
    ssc = (const float*)d_in[1];
    sw  = (const float*)d_in[2];
    x   = (const float*)d_in[n_in - 1];
    const int map[5] = {3, 1, 0, 4, 2};  // canonical -> sorted head index (dim,height,hm,reg,rot)
    for (int t = 0; t < 6; ++t)
      for (int h = 0; h < 5; ++h) {
        void* const* a = d_in + 3 + (t * 5 + map[h]) * 6;  // b1,b2,be1,s1,w1,w2
        b1p[t][h]  = (const float*)a[0]; w2p[t][h] = (const float*)a[5];
        b2p[t][h]  = (const float*)a[1]; be1p[t][h] = (const float*)a[2];
        s1p[t][h]  = (const float*)a[3]; w1p[t][h]  = (const float*)a[4];
      }
  }

  // workspace carve-up
  char* ws = (char*)d_ws;
  size_t used = 0;
  auto carve = [&](size_t sz) -> char* {
    char* p = ws + used;
    used += (sz + 255) & ~(size_t)255;
    return p;
  };
  half_t* xh    = (half_t*)carve((size_t)XN * 2);
  half_t* wsh   = (half_t*)carve((size_t)9 * 64 * 512 * 2);
  half_t* feat  = (half_t*)carve((size_t)MTOT_ * 64 * 2);
  half_t* wt1   = (half_t*)carve((size_t)9 * 320 * 64 * 2);
  float*  bnsc  = (float*)carve(320 * 4);
  float*  bnbi  = (float*)carve(320 * 4);
  half_t* fmid  = (half_t*)carve((size_t)MTOT_ * 320 * 2);
  float*  planes= (float*)carve((size_t)10 * MTOT_ * 4);
  float*  nmsb  = (float*)carve((size_t)8 * HWP_ * 4);
  float*  cs    = (float*)carve((size_t)BB_ * 2048 * 4);
  int*    ci    = (int*)carve((size_t)BB_ * 2048 * 4);
  if (used > ws_size) return;  // insufficient scratch; nothing safe to do

  // 1) input NHWC f16
  dim3 gT((WW_ + 31) / 32, 512 / 32, BB_ * HH_);
  nchw_to_nhwc_f2h<<<gT, 256, 0, stream>>>(x, xh, 512);
  // 2) shared conv weights
  prep_shared_w<<<(9 * 64 * 512 + 255) / 256, 256, 0, stream>>>(sw, wsh);
  // 3) shared conv (512->64) + BN + ReLU
  dim3 gS((MTOT_ + 255) / 256, 1);
  conv3x3_wmma<<<gS, 256, 0, stream>>>(xh, 512, wsh, 64, ssc, sbi, feat);

  const int ncls[6]  = {1, 2, 2, 1, 2, 2};
  const int lbase[6] = {1, 2, 4, 6, 7, 9};
  const int coutArr[5]  = {2, 1, 3, 2, 0};  // reg,height,dim,rot,(hm per-task)
  const int planeArr[5] = {2, 4, 5, 8, 0};  // hm occupies planes 0..1

  for (int t = 0; t < 6; ++t) {
    TaskW tw; TaskBN tb;
    for (int h = 0; h < 5; ++h) {
      tw.w1[h] = w1p[t][h]; tb.b1[h] = b1p[t][h];
      tb.s1[h] = s1p[t][h]; tb.be1[h] = be1p[t][h];
    }
    prep_task_w<<<(9 * 320 * 64 + 255) / 256, 256, 0, stream>>>(tw, wt1);
    prep_task_bn<<<5, 64, 0, stream>>>(tb, bnsc, bnbi);

    // fused 5-head conv1 (64->320) + BN + ReLU
    dim3 gH((MTOT_ + 255) / 256, 5);
    conv3x3_wmma<<<gH, 256, 0, stream>>>(feat, 64, wt1, 320, bnsc, bnbi, fmid);

    // per-head tiny conv2
    for (int h = 0; h < 5; ++h) {
      int cout = (h == 4) ? ncls[t] : coutArr[h];
      conv2_direct<<<(MTOT_ + 255) / 256, 256, 0, stream>>>(
          fmid, 320, h * 64, w2p[t][h], b2p[t][h],
          planes + (size_t)planeArr[h] * MTOT_, cout);
    }

    // decode
    int C = ncls[t];
    int tot = BB_ * C * HWP_;
    nms_sigmoid<<<(tot + 255) / 256, 256, 0, stream>>>(planes, C, nmsb);
    init_cand<<<(BB_ * 2 * 1024 + 255) / 256, 256, 0, stream>>>(cs, ci);
    topk_select<<<BB_ * C, 256, 0, stream>>>(nmsb, C, cs, ci);
    sort_and_emit<<<BB_, 512, 0, stream>>>(cs, ci, planes, C, t, lbase[t], (float*)d_out);
  }
  (void)out_size;
}
